// RGBuvHistBlock_77412490543607
// MI455X (gfx1250) — matrix operand order repaired and checked by reading, not compiled, not run on hardware
//
#include <hip/hip_runtime.h>

#define NPIX 22500
#define HB 64
#define LIMGS 32
#define KSPLIT 8
#define PIX_PER_BLOCK 2816   // 22 batches of 128; KSPLIT*2816 >= 22500
#define BATCH 128
#define PITCH 136            // 128 + 8 halves pad (rotates LDS banks per row)
#define EPS 1e-6f
#define INV_SIGMA 50.0f      // 1/0.02

typedef __attribute__((ext_vector_type(16))) _Float16 v16h;
typedef __attribute__((ext_vector_type(8)))  _Float16 v8h;
typedef __attribute__((ext_vector_type(2)))  _Float16 h2;
typedef __attribute__((ext_vector_type(8)))  float    v8f;

union Frag16 { v16h v; v8h h[2]; };

__global__ void zero_ws_kernel(float* __restrict__ ws, int n) {
    int i = blockIdx.x * blockDim.x + threadIdx.x;
    if (i < n) ws[i] = 0.0f;
}

__launch_bounds__(256)
__global__ void rgbuv_hist_kernel(const float* __restrict__ x, float* __restrict__ ws) {
    // LDS staging: A = w (u-bin major), B = kv (v-bin major), K = pixel within batch
    __shared__ _Float16 sA[HB][PITCH];
    __shared__ _Float16 sB[HB][PITCH];
    __shared__ float sIu[BATCH];
    __shared__ float sIv[BATCH];
    __shared__ float sIy[BATCH];

    const int ksp = blockIdx.x;            // K-split
    const int c   = blockIdx.y;            // channel
    const int l   = blockIdx.z;            // image
    const int c1  = (c == 2) ? 0 : c + 1;
    const int c2  = (c == 0) ? 2 : c - 1;  // (c+2)%3

    const float* __restrict__ xb = x + (size_t)l * 3 * NPIX;
    const int pixStart = ksp * PIX_PER_BLOCK;
    const int pixLimit = (pixStart + PIX_PER_BLOCK < NPIX) ? (pixStart + PIX_PER_BLOCK) : NPIX;

    const int tid  = threadIdx.x;
    const int lane = tid & 31;
    const int wid  = tid >> 5;             // 8 waves
    const int lr   = lane & 15;
    const int half = lane >> 4;

    // Each wave: one 16-row M strip (u), two 16-col N tiles (v)
    const int mTile = wid >> 1;            // 0..3
    const int nT0   = (wid & 1) * 2;       // 0 or 2 -> tiles nT0, nT0+1

    v8f acc0 = {};
    v8f acc1 = {};

    // phase-2 assignment: thread owns bin = tid&63, pixel-pair group pg = tid>>6
    const int   bin  = tid & 63;
    const int   pg   = tid >> 6;
    const float edge = -3.0f + (6.0f / 63.0f) * (float)bin;

    for (int base = pixStart; base < pixLimit; base += BATCH) {
        // ---- phase 1: per-pixel features (Iu, Iv, Iy) -> LDS ----
        if (tid < BATCH) {
            int p = base + tid;
            float iy = 0.0f, iu = 0.0f, iv = 0.0f;
            if (p < pixLimit) {
                float I0 = __builtin_fminf(__builtin_fmaxf(xb[p], 0.0f), 1.0f);
                float I1 = __builtin_fminf(__builtin_fmaxf(xb[NPIX + p], 0.0f), 1.0f);
                float I2 = __builtin_fminf(__builtin_fmaxf(xb[2 * NPIX + p], 0.0f), 1.0f);
                iy = __builtin_sqrtf(__builtin_fmaf(I0, I0, __builtin_fmaf(I1, I1, I2 * I2)) + EPS);
                float Ic  = (c  == 0) ? I0 : ((c  == 1) ? I1 : I2);
                float Ic1 = (c1 == 0) ? I0 : ((c1 == 1) ? I1 : I2);
                float Ic2 = (c2 == 0) ? I0 : ((c2 == 1) ? I1 : I2);
                float lc  = __logf(Ic  + EPS);
                float lc1 = __logf(Ic1 + EPS);
                float lc2 = __logf(Ic2 + EPS);
                iu = lc - lc1;
                iv = lc - lc2;
            }
            sIy[tid] = iy;   // iy==0 kills dead pixels (w = iy*ku = 0)
            sIu[tid] = iu;
            sIv[tid] = iv;
        }
        __syncthreads();

        // ---- phase 2: inverse-quadratic kernel eval, pack f16 pairs into LDS ----
#pragma unroll 4
        for (int j = 0; j < 16; ++j) {
            int p0 = pg * 2 + j * 8;       // even; covers all 128 pixels across pg
            float iu0 = sIu[p0], iu1 = sIu[p0 + 1];
            float iv0 = sIv[p0], iv1 = sIv[p0 + 1];
            float iy0 = sIy[p0], iy1 = sIy[p0 + 1];

            float t0 = (iu0 - edge) * INV_SIGMA;
            float t1 = (iu1 - edge) * INV_SIGMA;
            float w0 = iy0 * __builtin_amdgcn_rcpf(__builtin_fmaf(t0, t0, 1.0f));
            float w1 = iy1 * __builtin_amdgcn_rcpf(__builtin_fmaf(t1, t1, 1.0f));
            h2 wa; wa.x = (_Float16)w0; wa.y = (_Float16)w1;
            *(h2*)&sA[bin][p0] = wa;

            float s0 = (iv0 - edge) * INV_SIGMA;
            float s1 = (iv1 - edge) * INV_SIGMA;
            float k0 = __builtin_amdgcn_rcpf(__builtin_fmaf(s0, s0, 1.0f));
            float k1 = __builtin_amdgcn_rcpf(__builtin_fmaf(s1, s1, 1.0f));
            h2 kb; kb.x = (_Float16)k0; kb.y = (_Float16)k1;
            *(h2*)&sB[bin][p0] = kb;
        }
        __syncthreads();

        // ---- WMMA phase: 4 K-steps of 32 pixels ----
        const int arow  = mTile * 16 + lr;
        const int bcol0 = nT0 * 16 + lr;
        const int bcol1 = bcol0 + 16;
#pragma unroll
        for (int kstep = 0; kstep < 4; ++kstep) {
            const int kb = kstep * 32;
            // A 16x32 f16 layout: lanes 0-15 hold K kb+0..15, lanes 16-31 K kb+16..31 (matches B)
            Frag16 a;
            a.h[0] = *(const v8h*)&sA[arow][kb + half * 16];
            a.h[1] = *(const v8h*)&sA[arow][kb + half * 16 + 8];
            // B 32x16 f16 layout: lane = column; lanes 0-15 K kb+0..15, lanes 16-31 K kb+16..31
            Frag16 b0, b1;
            b0.h[0] = *(const v8h*)&sB[bcol0][kb + half * 16];
            b0.h[1] = *(const v8h*)&sB[bcol0][kb + half * 16 + 8];
            b1.h[0] = *(const v8h*)&sB[bcol1][kb + half * 16];
            b1.h[1] = *(const v8h*)&sB[bcol1][kb + half * 16 + 8];

            acc0 = __builtin_amdgcn_wmma_f32_16x16x32_f16(false, a.v, false, b0.v,
                                                          (short)0, acc0, false, false);
            acc1 = __builtin_amdgcn_wmma_f32_16x16x32_f16(false, a.v, false, b1.v,
                                                          (short)0, acc1, false, false);
        }
        __syncthreads();
    }

    // ---- writeback: atomic f32 add of partial 64x64 histogram ----
    float* __restrict__ hb = ws + ((size_t)(l * 3 + c) << 12);
    const int v0 = nT0 * 16 + lr;
#pragma unroll
    for (int r = 0; r < 8; ++r) {
        int u = mTile * 16 + r + half * 8;   // C/D layout: VGPR r -> M=r (lanes 0-15) / r+8
        unsafeAtomicAdd(&hb[(u << 6) + v0], acc0[r]);
        unsafeAtomicAdd(&hb[(u << 6) + v0 + 16], acc1[r]);
    }
}

__launch_bounds__(256)
__global__ void normalize_kernel(const float* __restrict__ ws, float* __restrict__ out) {
    __shared__ float red[256];
    const int l = blockIdx.x;
    const float* __restrict__ src = ws + (size_t)l * 3 * HB * HB;
    float* __restrict__ dst = out + (size_t)l * 3 * HB * HB;

    float s = 0.0f;
    for (int i = threadIdx.x; i < 3 * HB * HB; i += 256) s += src[i];
    red[threadIdx.x] = s;
    __syncthreads();
#pragma unroll
    for (int off = 128; off > 0; off >>= 1) {
        if (threadIdx.x < off) red[threadIdx.x] += red[threadIdx.x + off];
        __syncthreads();
    }
    const float inv = 1.0f / (red[0] + EPS);
    for (int i = threadIdx.x; i < 3 * HB * HB; i += 256) dst[i] = src[i] * inv;
}

extern "C" void kernel_launch(void* const* d_in, const int* in_sizes, int n_in,
                              void* d_out, int out_size, void* d_ws, size_t ws_size,
                              hipStream_t stream) {
    const float* x = (const float*)d_in[0];
    float* out = (float*)d_out;
    float* ws  = (float*)d_ws;     // 32*3*64*64 f32 accumulator = 1.5 MB

    const int nacc = LIMGS * 3 * HB * HB;
    zero_ws_kernel<<<(nacc + 255) / 256, 256, 0, stream>>>(ws, nacc);

    dim3 grid(KSPLIT, 3, LIMGS);   // 768 workgroups
    rgbuv_hist_kernel<<<grid, 256, 0, stream>>>(x, ws);

    normalize_kernel<<<LIMGS, 256, 0, stream>>>(ws, out);
}
